// GNBlockLite_86844238725710
// MI455X (gfx1250) — matrix operand order repaired and checked by reading, not compiled, not run on hardware
//
#include <hip/hip_runtime.h>
#include <hip/hip_bf16.h>

// ---------------------------------------------------------------------------
// GN block (GNBlockLite) for MI455X / gfx1250.
// Edge+node MLPs run as bf16 WMMA GEMMs (v_wmma_f32_16x16x32_bf16), wave32.
// Partitioning: each wave owns ALL 64 rows x one N-strip, so weight (B)
// fragments are loaded from L2 exactly once per WG and reused 4x in registers.
// ---------------------------------------------------------------------------

#define BDIM 64
#define NDIM 64
#define E_DIM 128
#define N_DIM 128
#define G_DIM 128
#define C_DIM 64
#define HHEADS 4
#define NEGV (-1e9f)

typedef __attribute__((ext_vector_type(16))) __bf16 v16bf;
typedef __attribute__((ext_vector_type(8)))  float  v8f;

union Frag16 { uint4 q[2]; v16bf v; };

static __device__ inline unsigned short f2bf(float f) {
    unsigned int u = __float_as_uint(f);
    u += 0x7FFFu + ((u >> 16) & 1u);          // round-to-nearest-even
    return (unsigned short)(u >> 16);
}

static __device__ inline v8f wmma_bf16(v16bf a, v16bf b, v8f c) {
    // (neg_a, A, neg_b, B, c_mod, C, reuse_a, reuse_b)
    return __builtin_amdgcn_wmma_f32_16x16x32_bf16(false, a, false, b, (short)0, c, false, false);
}

// A fragment: 16x32 bf16, row-major source (ld in shorts).
// lane<16: M=lane, elems 0..7 -> K=k0+0..7, 8..15 -> K=k0+16..23
// lane>=16: M=lane-16, elems 0..7 -> K=k0+8..15, 8..15 -> K=k0+24..31
static __device__ inline v16bf load_afrag(const unsigned short* base, int ld, int k0, int lane) {
    const int half = lane >> 4, m = lane & 15;
    const unsigned short* p = base + m * ld + k0 + half * 8;
    Frag16 f;
    f.q[0] = *(const uint4*)(p);
    f.q[1] = *(const uint4*)(p + 16);
    return f.v;
}

// B fragment: 32x16 bf16 from W^T[n][k] row-major (ld shorts per row).
// lane<16: N=n0+lane, elems 0..7 -> K=k0+0..7, 8..15 -> K=k0+16..23
// lane>=16: N=n0+lane-16, elems 0..7 -> K=k0+8..15, 8..15 -> K=k0+24..31
static __device__ inline v16bf load_bfrag(const unsigned short* WT, int ld, int n0, int k0, int lane) {
    const int half = lane >> 4, n = lane & 15;
    const unsigned short* p = WT + (n0 + n) * ld + k0 + half * 8;
    Frag16 f;
    f.q[0] = *(const uint4*)(p);
    f.q[1] = *(const uint4*)(p + 16);
    return f.v;
}

// ---------------------------------------------------------------------------
// Weight prep: dst[n][k] = bf16(src[k][n]) for n < srcN else 0 (pad).
// ---------------------------------------------------------------------------
__global__ void prep_wt(unsigned short* __restrict__ dst, const float* __restrict__ src,
                        int K, int srcN, int dstN) {
    int idx = blockIdx.x * 256 + threadIdx.x;
    if (idx >= dstN * K) return;
    int n = idx / K, k = idx - n * K;
    dst[n * K + k] = (n < srcN) ? f2bf(src[k * srcN + n]) : (unsigned short)0;
}

// ---------------------------------------------------------------------------
// Fused LN + dual-MLP (attn logits + features) over a 64-row tile.
//   EDGE=true : rows = 64 receivers for fixed (b, sender s); K1=576, LNK=384
//   EDGE=false: rows = 64 nodes for batch b;                K1=448, LNK=256
// Hidden layout: cols [0,256) = attn-MLP hidden, [256,512) = feat-MLP hidden.
// ---------------------------------------------------------------------------
template <bool EDGE, int K1, int LNK>
__launch_bounds__(256)
__global__ void fused_mlp64(
    const float* __restrict__ nodes, const float* __restrict__ edges,
    const float* __restrict__ globs, const unsigned char* __restrict__ mask,
    const float* __restrict__ ctxt,
    const float* __restrict__ ln_g, const float* __restrict__ ln_b,
    const unsigned short* __restrict__ WT1,   // [512][K1] bf16 (attn | feat)
    const float* __restrict__ b1a, const float* __restrict__ b1f,
    const unsigned short* __restrict__ WT2f,  // [128][256] bf16
    const float* __restrict__ b2f,
    const unsigned short* __restrict__ WT2a,  // [16][256] bf16 (cols>=4 zero)
    const float* __restrict__ b2a,
    const float* __restrict__ pooledE,        // node mode only: [B][64][128]
    float* __restrict__ outFeat,              // d_out edges / d_out nodes
    float* __restrict__ outLogit)             // ws logits [B,N,N,4] / nlog [B,N,4]
{
    __shared__ unsigned short ldsA[64 * K1];
    __shared__ unsigned short ldsH[64 * 512];
    __shared__ float red[64 * 8];
    __shared__ float mus[64 * 2];
    __shared__ float mk[64];

    const int tid  = threadIdx.x;
    const int lane = tid & 31, wave = tid >> 5;     // 8 waves
    const int half = lane >> 4, ln16 = lane & 15;

    int b, s;
    if (EDGE) { b = blockIdx.x >> 6; s = blockIdx.x & 63; }
    else      { b = blockIdx.x;      s = 0; }

    auto rawval = [&](int r, int k) -> float {
        if (EDGE) {
            if (k < 128) return nodes[(b * 64 + s) * 128 + k];            // sender
            if (k < 256) return nodes[(b * 64 + r) * 128 + (k - 128)];    // receiver
            return edges[((b * 64 + s) * 64 + r) * 128 + (k - 256)];
        } else {
            if (k < 128) return nodes[(b * 64 + r) * 128 + k];
            return pooledE[(b * 64 + r) * 128 + (k - 128)];
        }
    };

    // ---- LayerNorm statistics (4 threads per row) ----
    {
        const int r = tid >> 2, q = tid & 3;
        const int seg = LNK / 4;
        float s1 = 0.f, s2 = 0.f;
        for (int j = q * seg; j < (q + 1) * seg; ++j) {
            float v = rawval(r, j);
            s1 += v; s2 += v * v;
        }
        red[r * 8 + q]     = s1;
        red[r * 8 + 4 + q] = s2;
    }
    __syncthreads();
    if (tid < 64) {
        float s1 = red[tid*8] + red[tid*8+1] + red[tid*8+2] + red[tid*8+3];
        float s2 = red[tid*8+4] + red[tid*8+5] + red[tid*8+6] + red[tid*8+7];
        float mu  = s1 / (float)LNK;
        float var = s2 / (float)LNK - mu * mu;
        mus[tid*2]   = mu;
        mus[tid*2+1] = rsqrtf(var + 1e-5f);
        mk[tid] = (float)mask[b * 64 + tid];
    }
    __syncthreads();

    // ---- Build bf16 A-tile: [64][K1] = [ LN(x)*g+b | globs | ctxt ] ----
    for (int idx = tid; idx < 64 * K1; idx += 256) {
        int r = idx / K1, k = idx - r * K1;
        float x;
        if (k < LNK) {
            float v = rawval(r, k);
            x = (v - mus[r*2]) * mus[r*2+1] * ln_g[k] + ln_b[k];
        } else if (k < LNK + G_DIM) {
            x = globs[b * G_DIM + (k - LNK)];
        } else {
            x = ctxt[b * C_DIM + (k - LNK - G_DIM)];
        }
        ldsA[r * K1 + k] = f2bf(x);
    }
    __syncthreads();

    // ---- Layer 1: [64 x K1] @ WT1^T -> hidden [64 x 512], bias + LeakyReLU(0.1)
    // Wave owns all 64 rows x 64 cols [wave*64, wave*64+64): B frag reused 4x.
    {
        v8f acc[16];
        #pragma unroll
        for (int t = 0; t < 16; ++t)
            #pragma unroll
            for (int v = 0; v < 8; ++v) acc[t][v] = 0.f;

        for (int kc = 0; kc < K1 / 32; ++kc) {
            const int k0 = kc * 32;
            v16bf a0 = load_afrag(&ldsA[ 0 * K1], K1, k0, lane);
            v16bf a1 = load_afrag(&ldsA[16 * K1], K1, k0, lane);
            v16bf a2 = load_afrag(&ldsA[32 * K1], K1, k0, lane);
            v16bf a3 = load_afrag(&ldsA[48 * K1], K1, k0, lane);
            #pragma unroll
            for (int nt = 0; nt < 4; ++nt) {
                v16bf bb = load_bfrag(WT1, K1, wave * 64 + nt * 16, k0, lane);
                acc[0*4+nt] = wmma_bf16(a0, bb, acc[0*4+nt]);
                acc[1*4+nt] = wmma_bf16(a1, bb, acc[1*4+nt]);
                acc[2*4+nt] = wmma_bf16(a2, bb, acc[2*4+nt]);
                acc[3*4+nt] = wmma_bf16(a3, bb, acc[3*4+nt]);
            }
        }
        #pragma unroll
        for (int nt = 0; nt < 4; ++nt) {
            const int n = wave * 64 + nt * 16 + ln16;
            const float bias = (n < 256) ? b1a[n] : b1f[n - 256];
            #pragma unroll
            for (int mt = 0; mt < 4; ++mt) {
                #pragma unroll
                for (int v = 0; v < 8; ++v) {
                    float h = acc[mt*4+nt][v] + bias;
                    h = (h > 0.f) ? h : 0.1f * h;
                    ldsH[(mt * 16 + half * 8 + v) * 512 + n] = f2bf(h);
                }
            }
        }
    }
    __syncthreads();

    // ---- Layer 2 (feat): hidden[:,256:512] @ WT2f^T -> [64 x 128]
    // Wave owns all 64 rows x 16 cols [wave*16, wave*16+16): B frag reused 4x.
    {
        v8f accF[4];
        #pragma unroll
        for (int t = 0; t < 4; ++t)
            #pragma unroll
            for (int v = 0; v < 8; ++v) accF[t][v] = 0.f;
        for (int kc = 0; kc < 8; ++kc) {
            const int k0 = kc * 32;
            v16bf bb = load_bfrag(WT2f, 256, wave * 16, k0, lane);
            #pragma unroll
            for (int mt = 0; mt < 4; ++mt) {
                v16bf a = load_afrag(&ldsH[(mt * 16) * 512 + 256], 512, k0, lane);
                accF[mt] = wmma_bf16(a, bb, accF[mt]);
            }
        }
        const int n = wave * 16 + ln16;
        const float bias = b2f[n];
        #pragma unroll
        for (int mt = 0; mt < 4; ++mt) {
            #pragma unroll
            for (int v = 0; v < 8; ++v) {
                const int r = mt * 16 + half * 8 + v;
                float val = accF[mt][v] + bias;
                if (EDGE) {
                    const int gi = ((b * 64 + s) * 64 + r) * 128 + n;
                    outFeat[gi] = (val + edges[gi]) * (mk[s] * mk[r]);   // residual, masked
                } else {
                    const int gi = (b * 64 + r) * 128 + n;
                    outFeat[gi] = val * mk[r] + nodes[gi];               // residual
                }
            }
        }
    }

    // ---- Layer 2 (attn logits): hidden[:,0:256] @ WT2a^T -> [64 x 16] (4 real)
    if (wave < 4) {
        const int mt = wave;        // waves 0..3 each own one 16-row tile
        v8f accA;
        #pragma unroll
        for (int v = 0; v < 8; ++v) accA[v] = 0.f;
        for (int kc = 0; kc < 8; ++kc) {
            const int k0 = kc * 32;
            v16bf a = load_afrag(&ldsH[(mt * 16) * 512], 512, k0, lane);
            v16bf bb = load_bfrag(WT2a, 256, 0, k0, lane);
            accA = wmma_bf16(a, bb, accA);
        }
        if (ln16 < HHEADS) {
            const float bias = b2a[ln16];
            #pragma unroll
            for (int v = 0; v < 8; ++v) {
                const int r = mt * 16 + half * 8 + v;
                const float val = accA[v] + bias;
                if (EDGE) outLogit[((b * 64 + s) * 64 + r) * 4 + ln16] = val;
                else      outLogit[(b * 64 + r) * 4 + ln16] = val;
            }
        }
    }
}

// ---------------------------------------------------------------------------
// Edge pooling: per (b, receiver r): masked softmax over senders, weighted sum.
// ---------------------------------------------------------------------------
__global__ void edge_pool(const float* __restrict__ newEdges, const float* __restrict__ logits,
                          const unsigned char* __restrict__ mask, float* __restrict__ pooledE) {
    __shared__ float lw[64 * 4];
    const int b = blockIdx.x >> 6, r = blockIdx.x & 63;
    const int tid = threadIdx.x;  // 128
    const float mr = (float)mask[b * 64 + r];

    for (int idx = tid; idx < 256; idx += 128) {
        const int s = idx >> 2, h = idx & 3;
        const float adj = (float)mask[b * 64 + s] * mr;
        const float l = logits[((b * 64 + s) * 64 + r) * 4 + h];
        lw[idx] = (adj > 0.f) ? l : NEGV;
    }
    __syncthreads();
    if (tid < 4) {
        const int h = tid;
        float mx = -3.4e38f;
        for (int s = 0; s < 64; ++s) mx = fmaxf(mx, lw[s * 4 + h]);
        float sum = 0.f;
        for (int s = 0; s < 64; ++s) sum += __expf(lw[s * 4 + h] - mx);
        const float inv = 1.f / sum;
        for (int s = 0; s < 64; ++s) {
            const float adj = (float)mask[b * 64 + s] * mr;
            lw[s * 4 + h] = (adj > 0.f) ? __expf(lw[s * 4 + h] - mx) * inv : 0.f;
        }
    }
    __syncthreads();
    {
        const int e = tid, h = e >> 5;  // HEAD_E = 32
        float acc = 0.f;
        for (int s = 0; s < 64; ++s)
            acc += newEdges[((b * 64 + s) * 64 + r) * 128 + e] * lw[s * 4 + h];
        pooledE[(b * 64 + r) * 128 + e] = acc * 0.08838834764831845f /* 1/sqrt(128) */ * mr;
    }
}

// ---------------------------------------------------------------------------
// Node pooling: per batch b: masked softmax over nodes, weighted sum.
// ---------------------------------------------------------------------------
__global__ void node_pool(const float* __restrict__ newNodes, const float* __restrict__ nlog,
                          const unsigned char* __restrict__ mask, float* __restrict__ pooledN) {
    __shared__ float lw[64 * 4];
    const int b = blockIdx.x;
    const int tid = threadIdx.x;  // 128

    for (int idx = tid; idx < 256; idx += 128) {
        const int n = idx >> 2, h = idx & 3;
        const float m = (float)mask[b * 64 + n];
        lw[idx] = (m > 0.f) ? nlog[(b * 64 + n) * 4 + h] : NEGV;
    }
    __syncthreads();
    if (tid < 4) {
        const int h = tid;
        float mx = -3.4e38f;
        for (int n = 0; n < 64; ++n) mx = fmaxf(mx, lw[n * 4 + h]);
        float sum = 0.f;
        for (int n = 0; n < 64; ++n) sum += __expf(lw[n * 4 + h] - mx);
        const float inv = 1.f / sum;
        for (int n = 0; n < 64; ++n) {
            const float m = (float)mask[b * 64 + n];
            lw[n * 4 + h] = __expf(lw[n * 4 + h] - mx) * inv * 0.08838834764831845f * m;
        }
    }
    __syncthreads();
    {
        const int e = tid, h = e >> 5;  // HEAD_N = 32
        float acc = 0.f;
        for (int n = 0; n < 64; ++n)
            acc += newNodes[(b * 64 + n) * 128 + e] * lw[n * 4 + h];
        pooledN[b * 128 + e] = acc;
    }
}

// ---------------------------------------------------------------------------
// Glob block (tiny, scalar): LN(globs||pooled_nodes) -> MLP(320->256->128) + res
// ---------------------------------------------------------------------------
__global__ void glob_mlp(const float* __restrict__ globs, const float* __restrict__ pooledN,
                         const float* __restrict__ ctxt,
                         const float* __restrict__ ln_g, const float* __restrict__ ln_b,
                         const float* __restrict__ W1, const float* __restrict__ b1,
                         const float* __restrict__ W2, const float* __restrict__ b2,
                         float* __restrict__ outG) {
    __shared__ float xs[320];
    __shared__ float hs[256];
    __shared__ float st[2];
    const int b = blockIdx.x, tid = threadIdx.x;  // 256
    hs[tid] = (tid < 128) ? globs[b * 128 + tid] : pooledN[b * 128 + tid - 128];
    __syncthreads();
    if (tid == 0) {
        float s1 = 0.f, s2 = 0.f;
        for (int i = 0; i < 256; ++i) { float v = hs[i]; s1 += v; s2 += v * v; }
        float mu = s1 / 256.f, var = s2 / 256.f - mu * mu;
        st[0] = mu; st[1] = rsqrtf(var + 1e-5f);
    }
    __syncthreads();
    xs[tid] = (hs[tid] - st[0]) * st[1] * ln_g[tid] + ln_b[tid];
    if (tid < 64) xs[256 + tid] = ctxt[b * 64 + tid];
    __syncthreads();
    float acc = b1[tid];
    for (int k = 0; k < 320; ++k) acc += xs[k] * W1[k * 256 + tid];
    acc = (acc > 0.f) ? acc : 0.1f * acc;
    __syncthreads();
    hs[tid] = acc;
    __syncthreads();
    if (tid < 128) {
        float o = b2[tid];
        for (int k = 0; k < 256; ++k) o += hs[k] * W2[k * 128 + tid];
        outG[b * 128 + tid] = o + globs[b * 128 + tid];
    }
}

// ---------------------------------------------------------------------------
extern "C" void kernel_launch(void* const* d_in, const int* in_sizes, int n_in,
                              void* d_out, int out_size, void* d_ws, size_t ws_size,
                              hipStream_t stream) {
    (void)in_sizes; (void)n_in; (void)out_size; (void)ws_size;

    // setup_inputs() source order, params flattened in JAX pytree (sorted-key) order:
    // mlp -> W1, W2, b1, b2 ; ln -> b, g
    const float*         nodes = (const float*)d_in[0];          // [64,64,128]
    const float*         edges = (const float*)d_in[1];          // [64,64,64,128]
    const float*         globs = (const float*)d_in[2];          // [64,128]
    const unsigned char* mask  = (const unsigned char*)d_in[3];  // [64,64] bool
    const float*         ctxt  = (const float*)d_in[4];          // [64,64]
    const float* eaW1 = (const float*)d_in[5];   // edge_attn.W1 [576,256]
    const float* eaW2 = (const float*)d_in[6];   // edge_attn.W2 [256,4]
    const float* eab1 = (const float*)d_in[7];
    const float* eab2 = (const float*)d_in[8];
    const float* efW1 = (const float*)d_in[9];   // edge_feat.W1 [576,256]
    const float* efW2 = (const float*)d_in[10];  // edge_feat.W2 [256,128]
    const float* efb1 = (const float*)d_in[11];
    const float* efb2 = (const float*)d_in[12];
    const float* elnb = (const float*)d_in[13];  // edge_ln.b [384]
    const float* elng = (const float*)d_in[14];  // edge_ln.g [384]
    const float* gfW1 = (const float*)d_in[15];  // glob_feat.W1 [320,256]
    const float* gfW2 = (const float*)d_in[16];  // glob_feat.W2 [256,128]
    const float* gfb1 = (const float*)d_in[17];
    const float* gfb2 = (const float*)d_in[18];
    const float* glnb = (const float*)d_in[19];  // glob_ln.b [256]
    const float* glng = (const float*)d_in[20];  // glob_ln.g [256]
    const float* naW1 = (const float*)d_in[21];  // node_attn.W1 [448,256]
    const float* naW2 = (const float*)d_in[22];  // node_attn.W2 [256,4]
    const float* nab1 = (const float*)d_in[23];
    const float* nab2 = (const float*)d_in[24];
    const float* nfW1 = (const float*)d_in[25];  // node_feat.W1 [448,256]
    const float* nfW2 = (const float*)d_in[26];  // node_feat.W2 [256,128]
    const float* nfb1 = (const float*)d_in[27];
    const float* nfb2 = (const float*)d_in[28];
    const float* nlnb = (const float*)d_in[29];  // node_ln.b [256]
    const float* nlng = (const float*)d_in[30];  // node_ln.g [256]

    // ---- workspace carve-out ----
    char* ws = (char*)d_ws;
    unsigned short* WT1e  = (unsigned short*)ws; ws += (size_t)512 * 576 * 2;
    unsigned short* WT1n  = (unsigned short*)ws; ws += (size_t)512 * 448 * 2;
    unsigned short* WT2fe = (unsigned short*)ws; ws += (size_t)128 * 256 * 2;
    unsigned short* WT2ae = (unsigned short*)ws; ws += (size_t)16  * 256 * 2;
    unsigned short* WT2fn = (unsigned short*)ws; ws += (size_t)128 * 256 * 2;
    unsigned short* WT2an = (unsigned short*)ws; ws += (size_t)16  * 256 * 2;
    float* wsLogit   = (float*)ws; ws += (size_t)64 * 64 * 64 * 4 * 4;  // edge logits
    float* wsPooledE = (float*)ws; ws += (size_t)64 * 64 * 128 * 4;
    float* wsNlog    = (float*)ws; ws += (size_t)64 * 64 * 4 * 4;
    float* wsPooledN = (float*)ws; ws += (size_t)64 * 128 * 4;

    float* outEdges = (float*)d_out;
    float* outNodes = outEdges + (size_t)64 * 64 * 64 * 128;
    float* outGlobs = outNodes + (size_t)64 * 64 * 128;

    // ---- weight prep (bf16 transposed layouts) ----
    auto prep = [&](unsigned short* dst, const float* src, int K, int srcN, int dstN) {
        const int tot = dstN * K;
        prep_wt<<<(tot + 255) / 256, 256, 0, stream>>>(dst, src, K, srcN, dstN);
    };
    prep(WT1e,             eaW1, 576, 256, 256);
    prep(WT1e + 256 * 576, efW1, 576, 256, 256);
    prep(WT1n,             naW1, 448, 256, 256);
    prep(WT1n + 256 * 448, nfW1, 448, 256, 256);
    prep(WT2fe, efW2, 256, 128, 128);
    prep(WT2ae, eaW2, 256,   4,  16);
    prep(WT2fn, nfW2, 256, 128, 128);
    prep(WT2an, naW2, 256,   4,  16);

    // ---- edge block: one WG per (b, sender) ----
    fused_mlp64<true, 576, 384><<<64 * 64, 256, 0, stream>>>(
        nodes, edges, globs, mask, ctxt, elng, elnb,
        WT1e, eab1, efb1, WT2fe, efb2, WT2ae, eab2,
        nullptr, outEdges, wsLogit);

    edge_pool<<<64 * 64, 128, 0, stream>>>(outEdges, wsLogit, mask, wsPooledE);

    // ---- node block: one WG per batch ----
    fused_mlp64<false, 448, 256><<<64, 256, 0, stream>>>(
        nodes, edges, globs, mask, ctxt, nlng, nlnb,
        WT1n, nab1, nfb1, WT2fn, nfb2, WT2an, nab2,
        wsPooledE, outNodes, wsNlog);

    node_pool<<<64, 128, 0, stream>>>(outNodes, wsNlog, mask, wsPooledN);

    // ---- glob block ----
    glob_mlp<<<64, 256, 0, stream>>>(globs, wsPooledN, ctxt, glng, glnb,
                                     gfW1, gfb1, gfW2, gfb2, outGlobs);
}
